// Abcnn2Portion_11785390260225
// MI455X (gfx1250) — matrix operand order repaired and checked by reading, not compiled, not run on hardware
//
#include <hip/hip_runtime.h>
#include <math.h>

// Problem constants from the reference
#define B_   16
#define L_   256
#define W_   4
#define H_   128
#define N_   259          // L + W - 1
#define NP_  272          // N padded to multiple of 16
#define NT_  17           // NP_/16 tiles
#define EPS_ 1e-6f

typedef __attribute__((ext_vector_type(16))) _Float16 v16h;
typedef __attribute__((ext_vector_type(8)))  _Float16 v8h;
typedef __attribute__((ext_vector_type(8)))  float    v8f;

// ---------------------------------------------------------------------------
// Kernel 1: stage x1/x2 as f16 (zero-padded to NP_ rows) and compute fp32
// per-row squared norms.  One block per (b, n); 128 threads = one h each.
// ---------------------------------------------------------------------------
__global__ void prep_kernel(const float* __restrict__ x1,
                            const float* __restrict__ x2,
                            _Float16* __restrict__ x1h,
                            _Float16* __restrict__ x2h,
                            float* __restrict__ n1,
                            float* __restrict__ n2) {
    int blk = blockIdx.x;            // b*NP_ + n
    int b = blk / NP_;
    int n = blk % NP_;
    int h = threadIdx.x;             // 0..127

    float v1 = 0.0f, v2 = 0.0f;
    if (n < N_) {
        size_t src = ((size_t)b * N_ + n) * H_ + h;
        v1 = x1[src];
        v2 = x2[src];
    }
    size_t dst = ((size_t)b * NP_ + n) * H_ + h;
    x1h[dst] = (_Float16)v1;
    x2h[dst] = (_Float16)v2;

    __shared__ float s1[H_];
    __shared__ float s2[H_];
    s1[h] = v1 * v1;
    s2[h] = v2 * v2;
    __syncthreads();
#pragma unroll
    for (int off = H_ / 2; off > 0; off >>= 1) {
        if (h < off) { s1[h] += s1[h + off]; s2[h] += s2[h + off]; }
        __syncthreads();
    }
    if (h == 0) { n1[blk] = s1[0]; n2[blk] = s2[0]; }
}

// ---------------------------------------------------------------------------
// Kernel 2: batched Gram matrix via v_wmma_f32_16x16x32_f16, fused with the
// attention transform and the row/column reductions.
// One wave per (b, i_tile); sweeps all NT_ j tiles with the next tile's B
// fragments prefetched (double-buffered) so WMMA + epilogue VALU hide the
// global-load latency of the following tile.
// ---------------------------------------------------------------------------
__global__ __launch_bounds__(32)
void att_wmma_kernel(const _Float16* __restrict__ x1h,
                     const _Float16* __restrict__ x2h,
                     const float* __restrict__ n1,
                     const float* __restrict__ n2,
                     float* __restrict__ p1,     // [B][NT_][NP_]
                     float* __restrict__ a2) {   // [B][NP_]
    int blk = blockIdx.x;            // b*NT_ + it
    int b  = blk / NT_;
    int it = blk % NT_;
    int i0 = it * 16;

    int lane = threadIdx.x;          // 0..31 (wave32)
    int m  = lane & 15;              // position within tile
    int hs = lane >> 4;              // half-wave select

    // --- A fragments: rows of x2 (M = i dim), K = 128 in 4 chunks of 32 ---
    const _Float16* aRow = x2h + ((size_t)b * NP_ + (i0 + m)) * H_;
    v16h afrag[4];
#pragma unroll
    for (int c = 0; c < 4; ++c) {
        int k0 = c * 32 + hs * 8;                   // VGPRs 0-3: K=hs*8+0..7
        v8h lo = *(const v8h*)(aRow + k0);
        v8h hi = *(const v8h*)(aRow + k0 + 16);     // VGPRs 4-7: K=16+hs*8+0..7
        afrag[c] = __builtin_shufflevector(lo, hi,
            0,1,2,3,4,5,6,7,8,9,10,11,12,13,14,15);
    }

    // per-lane constant x2 norms for the 8 D rows this lane touches
    float ni[8];
#pragma unroll
    for (int v = 0; v < 8; ++v) ni[v] = n2[b * NP_ + i0 + v + hs * 8];

    float racc[8];                   // row-sum accumulators (over all j)
#pragma unroll
    for (int v = 0; v < 8; ++v) racc[v] = 0.0f;

    const _Float16* x1base = x1h + (size_t)b * NP_ * H_;
    const float*    n1row  = n1 + b * NP_;
    float* p1row = p1 + ((size_t)b * NT_ + it) * NP_;

    // B fragment loader: lane = N, lo holds K hs*8+0..7,
    // hi holds K 16+hs*8+0..7 (same order as A).
    auto loadB = [&](int jt, v8h (&lo)[4], v8h (&hi)[4]) {
        const _Float16* bRow = x1base + (size_t)(jt * 16 + m) * H_;
#pragma unroll
        for (int c = 0; c < 4; ++c) {
            int k0 = c * 32 + hs * 8;
            lo[c] = *(const v8h*)(bRow + k0);
            hi[c] = *(const v8h*)(bRow + k0 + 16);
        }
    };

    // WMMA + attention epilogue for one 16x16 tile
    auto compute = [&](int jt, v8h (&lo)[4], v8h (&hi)[4]) {
        v8f acc = {};
#pragma unroll
        for (int c = 0; c < 4; ++c) {
            v16h bfrag = __builtin_shufflevector(lo[c], hi[c],
                0,1,2,3,4,5,6,7,8,9,10,11,12,13,14,15);
            acc = __builtin_amdgcn_wmma_f32_16x16x32_f16(
                false, afrag[c], false, bfrag, (short)0, acc, false, false);
        }
        int   j  = jt * 16 + m;
        float nj = n1row[j];
        float colsum = 0.0f;
#pragma unroll
        for (int v = 0; v < 8; ++v) {
            int   i  = i0 + v + hs * 8;
            float d2 = fmaxf(ni[v] + nj - 2.0f * acc[v], 0.0f);
            float att = 1.0f / (sqrtf(d2 + EPS_) + 1.0f);
            att = (i < N_ && j < N_) ? att : 0.0f;
            racc[v] += att;
            colsum  += att;
        }
        colsum += __shfl_xor(colsum, 16);   // join half-waves (rows 0-7 / 8-15)
        if (lane < 16) p1row[j] = colsum;
    };

    // Software pipeline: double-buffered B fragments, unrolled by 2 so the
    // buffers stay in fixed registers (no dynamic indexing -> no scratch).
    v8h loA[4], hiA[4], loB[4], hiB[4];
    loadB(0, loA, hiA);
#pragma unroll 1
    for (int jt = 0; jt < NT_ - 1; jt += 2) {     // jt = 0,2,...,14
        loadB(jt + 1, loB, hiB);                  // prefetch tile jt+1
        compute(jt, loA, hiA);                    // overlaps the loads above
        loadB(jt + 2, loA, hiA);                  // prefetch tile jt+2 (<=16)
        compute(jt + 1, loB, hiB);
    }
    compute(NT_ - 1, loA, hiA);                   // tile 16 (loaded last iter)

    // finish row sums: reduce across the 16 lanes of each half-wave
#pragma unroll
    for (int v = 0; v < 8; ++v) {
        float s = racc[v];
        s += __shfl_xor(s, 1);
        s += __shfl_xor(s, 2);
        s += __shfl_xor(s, 4);
        s += __shfl_xor(s, 8);
        if (m == 0)
            a2[b * NP_ + i0 + v + hs * 8] = s;   // unique writer -> no atomics
    }
}

// ---------------------------------------------------------------------------
// Kernel 3: reduce column partials over the 17 i-tiles -> a1[b][j].
// ---------------------------------------------------------------------------
__global__ void colsum_reduce_kernel(const float* __restrict__ p1,
                                     float* __restrict__ a1) {
    int idx = blockIdx.x * blockDim.x + threadIdx.x;   // b*NP_ + j
    if (idx >= B_ * NP_) return;
    int b = idx / NP_;
    int j = idx % NP_;
    float s = 0.0f;
#pragma unroll
    for (int it = 0; it < NT_; ++it)
        s += p1[((size_t)b * NT_ + it) * NP_ + j];
    a1[idx] = s;
}

// ---------------------------------------------------------------------------
// Kernel 4: fused weighted window-4 pooling epilogue for both outputs.
// out1[b,l,h] = sum_k x1[b,l+k,h]*a1[b,l+k]; out2 likewise with x2/a2.
// ---------------------------------------------------------------------------
__global__ void wp_kernel(const float* __restrict__ x1,
                          const float* __restrict__ x2,
                          const float* __restrict__ a1,
                          const float* __restrict__ a2,
                          float* __restrict__ out) {
    int idx = blockIdx.x * blockDim.x + threadIdx.x;
    if (idx >= B_ * L_ * H_) return;
    int h = idx % H_;
    int t = idx / H_;
    int l = t % L_;
    int b = t / L_;

    float o1 = 0.0f, o2 = 0.0f;
#pragma unroll
    for (int k = 0; k < W_; ++k) {
        int n = l + k;                                  // n <= 258 < N_
        size_t src = ((size_t)b * N_ + n) * H_ + h;
        o1 += x1[src] * a1[b * NP_ + n];
        o2 += x2[src] * a2[b * NP_ + n];
    }
    out[idx] = o1;
    out[(size_t)B_ * L_ * H_ + idx] = o2;
}

// ---------------------------------------------------------------------------
extern "C" void kernel_launch(void* const* d_in, const int* in_sizes, int n_in,
                              void* d_out, int out_size, void* d_ws, size_t ws_size,
                              hipStream_t stream) {
    (void)in_sizes; (void)n_in; (void)out_size; (void)ws_size;
    const float* x1 = (const float*)d_in[0];
    const float* x2 = (const float*)d_in[1];
    float* out = (float*)d_out;

    // workspace carve-up (~2.6 MB total)
    char* ws = (char*)d_ws;
    size_t off = 0;
    _Float16* x1h = (_Float16*)(ws + off); off += (size_t)B_ * NP_ * H_ * sizeof(_Float16);
    _Float16* x2h = (_Float16*)(ws + off); off += (size_t)B_ * NP_ * H_ * sizeof(_Float16);
    float* n1 = (float*)(ws + off); off += (size_t)B_ * NP_ * sizeof(float);
    float* n2 = (float*)(ws + off); off += (size_t)B_ * NP_ * sizeof(float);
    float* a1 = (float*)(ws + off); off += (size_t)B_ * NP_ * sizeof(float);
    float* a2 = (float*)(ws + off); off += (size_t)B_ * NP_ * sizeof(float);
    float* p1 = (float*)(ws + off); off += (size_t)B_ * NT_ * NP_ * sizeof(float);

    // 1) stage f16 + norms
    prep_kernel<<<B_ * NP_, H_, 0, stream>>>(x1, x2, x1h, x2h, n1, n2);

    // 2) WMMA attention tiles + fused reductions (one wave per 16-row strip)
    att_wmma_kernel<<<B_ * NT_, 32, 0, stream>>>(x1h, x2h, n1, n2, p1, a2);

    // 3) column-sum reduce
    colsum_reduce_kernel<<<(B_ * NP_ + 255) / 256, 256, 0, stream>>>(p1, a1);

    // 4) fused windowed-pool epilogue
    wp_kernel<<<(B_ * L_ * H_ + 255) / 256, 256, 0, stream>>>(x1, x2, a1, a2, out);
}
